// BoundaryConvLayer_7619271983310
// MI455X (gfx1250) — matrix operand order repaired and checked by reading, not compiled, not run on hardware
//
#include <hip/hip_runtime.h>
#include <hip/hip_bf16.h>

typedef __attribute__((ext_vector_type(16))) _Float16 v16h;
typedef __attribute__((ext_vector_type(8)))  float    v8f;

#define D64 64
#define EPS 1e-8f

// ---------------------------------------------------------------------------
// Kernel 0: zero the scatter accumulator S
// ---------------------------------------------------------------------------
__global__ void bc_zero_kernel(float* __restrict__ p, unsigned long long n) {
    unsigned long long tid = (unsigned long long)blockIdx.x * blockDim.x + threadIdx.x;
    if (tid < n) p[tid] = 0.0f;
}

// ---------------------------------------------------------------------------
// Kernel 1: pack the 4 weight matrices (f32 row-major [j][k]) into f16 tiles
// laid out exactly as WMMA B operands:
//   Wp[(((mat*4 + nt)*2 + kc)*32 + lane)*16 + i]  (i = 0..15 halfs, 32B/lane)
// where lane j' = lane&15 -> j = nt*16 + j', koff = (lane<16 ? 0 : 16),
// element i  ->  K = kc*32 + koff + i,  value = W[j][K]   (B = W^T).
// Total: 4*4*2*32 = 1024 lane-groups of 16 halfs = 32 KB.
// ---------------------------------------------------------------------------
__global__ void bc_pack_weights_kernel(
    const float* __restrict__ dir_w, const float* __restrict__ neu_w,
    const float* __restrict__ rob_w, const float* __restrict__ fc_w,
    _Float16* __restrict__ Wp)
{
    const int g = blockIdx.x * blockDim.x + threadIdx.x;   // 0..1023
    if (g >= 1024) return;
    const int lane = g & 31;
    const int kc   = (g >> 5) & 1;
    const int nt   = (g >> 6) & 3;
    const int mat  = (g >> 8) & 3;

    const float* Ws[4] = { dir_w, neu_w, rob_w, fc_w };
    const int j    = nt * 16 + (lane & 15);
    const int koff = kc * 32 + ((lane < 16) ? 0 : 16);
    const float* wrow = Ws[mat] + (size_t)j * D64 + koff;
    _Float16* dst = Wp + (size_t)g * 16;
    #pragma unroll
    for (int i = 0; i < 16; ++i) dst[i] = (_Float16)wrow[i];
}

// ---------------------------------------------------------------------------
// Kernel 2: fused 4x GEMM (WMMA f16 -> f32 acc) + boundary-coefficient math.
//   Produces: h = x@fc_w.T + fc_b
//             P = relu(x@neu_w.T+neu_b) / denom
//             Q (-> d_out) = (x@rob_w.T+rob_b) / denom
//   denom = relu(x@dir_w.T+dir_b) + relu(x@neu_w.T+neu_b)*degree + EPS
// One wave per 16-row tile of x; 8 waves per block. B comes pre-packed (f16).
// ---------------------------------------------------------------------------
__global__ void __launch_bounds__(256) bc_gemm_kernel(
    const float* __restrict__ x,
    const float* __restrict__ degree,
    const _Float16* __restrict__ Wp,
    const float* __restrict__ dir_b, const float* __restrict__ neu_b,
    const float* __restrict__ rob_b, const float* __restrict__ fc_b,
    float* __restrict__ h_out, float* __restrict__ P_out, float* __restrict__ Q_out,
    int n_tiles, int N)
{
    const int wave = threadIdx.x >> 5;
    const int lane = threadIdx.x & 31;
    const int tile = blockIdx.x * 8 + wave;
    if (tile >= n_tiles) return;            // wave-uniform: EXEC stays all-ones

    const int lhalf     = (lane < 16) ? 0 : 1;
    const int mrow_base = tile * 16;
    int m_lane = mrow_base + (lane & 15);
    if (m_lane >= N) m_lane = N - 1;        // clamp (N%16==0 in practice)

    // ---- Load A tiles (x rows), f32 -> f16, contiguous k order (k = 16h + s), same as packed B ----
    // lanes 0-15 : K = {0..15} (+32*kc); lanes 16-31: K = {16..31}
    v16h a[2];
    {
        const float* xr = x + (size_t)m_lane * D64;
        #pragma unroll
        for (int kc = 0; kc < 2; ++kc) {
            const int s0 = kc * 32 + lhalf * 16;
            #pragma unroll
            for (int i = 0; i < 8; ++i) {
                a[kc][i]     = (_Float16)xr[s0 + i];
                a[kc][8 + i] = (_Float16)xr[s0 + 8 + i];
            }
        }
    }

    // degree for the 8 accumulator rows this lane owns
    float deg[8];
    #pragma unroll
    for (int r = 0; r < 8; ++r) {
        int n = mrow_base + r + lhalf * 8;
        deg[r] = degree[(n < N) ? n : (N - 1)];
    }

    const float* Bs[4] = { dir_b, neu_b, rob_b, fc_b };
    const v16h* Wpv = (const v16h*)Wp;      // one v16h (32B) per lane-group

    #pragma unroll
    for (int nt = 0; nt < 4; ++nt) {
        const int j = nt * 16 + (lane & 15);   // output feature column
        v8f acc[4];

        #pragma unroll
        for (int mat = 0; mat < 4; ++mat) {
            v8f c = {0.f, 0.f, 0.f, 0.f, 0.f, 0.f, 0.f, 0.f};
            #pragma unroll
            for (int kc = 0; kc < 2; ++kc) {
                const v16h b = Wpv[((mat * 4 + nt) * 2 + kc) * 32 + lane];
                c = __builtin_amdgcn_wmma_f32_16x16x32_f16(
                        false, a[kc], false, b, (short)0, c, false, false);
            }
            const float bias = Bs[mat][j];
            #pragma unroll
            for (int r = 0; r < 8; ++r) acc[mat][r] = c[r] + bias;
        }

        // combine: alpha=relu(dir), beta=relu(neu), gamma=rob, h=fc
        #pragma unroll
        for (int r = 0; r < 8; ++r) {
            const float al = fmaxf(acc[0][r], 0.0f);
            const float be = fmaxf(acc[1][r], 0.0f);
            const float ga = acc[2][r];
            const float hh = acc[3][r];
            const float rden = 1.0f / (al + be * deg[r] + EPS);
            const int n = mrow_base + r + lhalf * 8;
            if (n < N) {
                const size_t off = (size_t)n * D64 + j;
                h_out[off] = hh;
                P_out[off] = be * rden;
                Q_out[off] = ga * rden;
            }
        }
    }
}

// ---------------------------------------------------------------------------
// Kernel 3: per-edge scatter: S[row] += h[col]  (32 lanes/edge, 2 floats/lane)
// h and S both fit in the 192 MB L2, so this runs at L2/atomic speed.
// ---------------------------------------------------------------------------
__global__ void __launch_bounds__(256) bc_scatter_kernel(
    const int* __restrict__ row, const int* __restrict__ col,
    const float* __restrict__ h, float* __restrict__ S, int E)
{
    const unsigned long long tid =
        (unsigned long long)blockIdx.x * blockDim.x + threadIdx.x;
    const unsigned long long e = tid >> 5;
    if (e >= (unsigned long long)E) return;
    const int d  = (int)(tid & 31);
    const int r  = row[e];
    const int cc = col[e];
    const float* hc = h + (size_t)cc * D64;
    float*       Sr = S + (size_t)r  * D64;
    atomicAdd(&Sr[d],      hc[d]);
    atomicAdd(&Sr[d + 32], hc[d + 32]);
}

// ---------------------------------------------------------------------------
// Kernel 4: out = P * (S + degree*h) + Q   (Q already resident in d_out)
// ---------------------------------------------------------------------------
__global__ void __launch_bounds__(256) bc_final_kernel(
    const float* __restrict__ P, const float* __restrict__ h,
    const float* __restrict__ S, const float* __restrict__ degree,
    float* __restrict__ out, int N)
{
    const unsigned long long tid =
        (unsigned long long)blockIdx.x * blockDim.x + threadIdx.x;
    const unsigned long long total = (unsigned long long)N * D64;
    if (tid >= total) return;
    const int n = (int)(tid >> 6);
    out[tid] = P[tid] * (S[tid] + degree[n] * h[tid]) + out[tid];
}

// ---------------------------------------------------------------------------
extern "C" void kernel_launch(void* const* d_in, const int* in_sizes, int n_in,
                              void* d_out, int out_size, void* d_ws, size_t ws_size,
                              hipStream_t stream)
{
    const float* x      = (const float*)d_in[0];
    const int*   ei     = (const int*)  d_in[1];   // [2, E] flat
    const float* degree = (const float*)d_in[2];
    const float* fc_w   = (const float*)d_in[3];
    const float* fc_b   = (const float*)d_in[4];
    const float* dir_w  = (const float*)d_in[5];
    const float* dir_b  = (const float*)d_in[6];
    const float* neu_w  = (const float*)d_in[7];
    const float* neu_b  = (const float*)d_in[8];
    const float* rob_w  = (const float*)d_in[9];
    const float* rob_b  = (const float*)d_in[10];

    const int N = in_sizes[0] / D64;
    const int E = in_sizes[1] / 2;
    const unsigned long long nd = (unsigned long long)N * D64;

    // workspace: h | P | S | packed f16 weights (32 KB)
    float* h = (float*)d_ws;
    float* P = h + nd;
    float* S = P + nd;
    _Float16* Wp = (_Float16*)(S + nd);
    float* out = (float*)d_out;

    // 0) zero scatter accumulator (every call; graph-replay safe)
    bc_zero_kernel<<<(unsigned)((nd + 255) / 256), 256, 0, stream>>>(S, nd);

    // 1) pack weights into WMMA-B f16 layout (tiny; stays hot in cache)
    bc_pack_weights_kernel<<<4, 256, 0, stream>>>(dir_w, neu_w, rob_w, fc_w, Wp);

    // 2) fused WMMA GEMMs -> h, P, Q(=d_out)
    const int n_tiles = (N + 15) / 16;
    bc_gemm_kernel<<<(n_tiles + 7) / 8, 256, 0, stream>>>(
        x, degree, Wp, dir_b, neu_b, rob_b, fc_b,
        h, P, out, n_tiles, N);

    // 3) edge scatter: S[row] += h[col]
    const unsigned long long sthreads = (unsigned long long)E * 32ull;
    bc_scatter_kernel<<<(unsigned)((sthreads + 255) / 256), 256, 0, stream>>>(
        ei, ei + E, h, S, E);

    // 4) out = P*(S + degree*h) + Q
    bc_final_kernel<<<(unsigned)((nd + 255) / 256), 256, 0, stream>>>(
        P, h, S, degree, out, N);
}
